// FirstPrinciplesAttention_6657199309530
// MI455X (gfx1250) — compile-verified, matrix operand order checked by reading, not run on hardware
//
#include <hip/hip_runtime.h>
#include <hip/hip_bf16.h>
#include <math.h>

typedef _Float16 half_t;
typedef __attribute__((ext_vector_type(16))) _Float16 v16h;
typedef __attribute__((ext_vector_type(8)))  float    v8f;

#define HEPS 1e-6f

// ---------------------------------------------------------------------------
// CDNA5 async DMA: global -> LDS, 16 bytes per lane. Tracked by ASYNCcnt.
// LDS operand is the wave-relative LDS byte offset (low 32 bits of the
// generic shared address).
// ---------------------------------------------------------------------------
static __device__ __forceinline__ void async_ld16(void* lds, const void* gaddr) {
  uint32_t lo = (uint32_t)(uintptr_t)lds;
  asm volatile("global_load_async_to_lds_b128 %0, %1, off"
               :: "v"(lo), "v"(gaddr)
               : "memory");
}
static __device__ __forceinline__ void async_wait0() {
  asm volatile("s_wait_asynccnt 0" ::: "memory");
}

// ---------------------------------------------------------------------------
// Fragment loader for V_WMMA_F32_16X16X32_F16 (gfx1250, wave32).
// A 16x32 f16 (ISA 7.12.2): lane L holds row m=L&15; lanes 0-15 carry
// K={0..7,16..23}, lanes 16-31 carry K={8..15,24..31}; VGPR v holds the pair
// k0 = (v>>2)*16 + kbase + (v&3)*2.  B 32x16 is symmetric with r = column n.
// Source is a row-major LDS tile base[r*ld + k]; pairs merge to ds_load_b128.
// ---------------------------------------------------------------------------
static __device__ __forceinline__ v16h frag_ld(const half_t* __restrict__ base, int ld) {
  const int lane = threadIdx.x & 31;
  const int r  = lane & 15;
  const int kb = (lane < 16) ? 0 : 8;
  v16h f;
#pragma unroll
  for (int v = 0; v < 8; ++v) {
    const int k0 = ((v >> 2) << 4) + kb + ((v & 3) << 1);
    union { float f32; half_t h[2]; } u;
    u.f32 = *(const float*)(base + r * ld + k0);
    f[2 * v]     = u.h[0];
    f[2 * v + 1] = u.h[1];
  }
  return f;
}

static __device__ __forceinline__ v8f wmma_f16(v16h a, v16h b, v8f c) {
  return __builtin_amdgcn_wmma_f32_16x16x32_f16(false, a, false, b, (short)0, c,
                                                false, false);
}

// ---------------------------------------------------------------------------
// Lorentz log-map at mu=(1,0,...,0) + (x - mu), emitted as f16 activations.
// ---------------------------------------------------------------------------
__global__ __launch_bounds__(256) void prep_kernel(const float* __restrict__ x,
                                                   half_t* __restrict__ vout,
                                                   half_t* __restrict__ xcout,
                                                   int d) {
  __shared__ float red[256];
  const int row = blockIdx.x;
  const int tid = threadIdx.x;
  const float* xr = x + (size_t)row * d;
  float sp[4];
  float ss = 0.f;
#pragma unroll
  for (int i = 0; i < 4; ++i) {
    int j = tid + i * 256;
    float xv = xr[j];
    float s = (j == 0) ? 0.f : fminf(fmaxf(xv, -8.f), 8.f);
    sp[i] = s;
    ss += s * s;
  }
  red[tid] = ss;
  __syncthreads();
  for (int off = 128; off > 0; off >>= 1) {
    if (tid < off) red[tid] += red[tid + off];
    __syncthreads();
  }
  const float ssum = red[0];
  const float y0 = sqrtf(1.f + ssum + HEPS);
  const float xy = fminf(-y0, -(1.f + HEPS));
  const float dd = fmaxf(acoshf(fmaxf(-xy, 1.f + HEPS)), 0.001f);
  const float dir0 = y0 + xy;
  const float dn = sqrtf(fmaxf(-dir0 * dir0 + ssum, HEPS));
  const float sc = dd / (dn + HEPS);
#pragma unroll
  for (int i = 0; i < 4; ++i) {
    int j = tid + i * 256;
    float vj = (j == 0) ? dir0 * sc : sp[i] * sc;
    vout[(size_t)row * d + j] = (half_t)vj;
    float xcj = xr[j] - ((j == 0) ? 1.f : 0.f);
    xcout[(size_t)row * d + j] = (half_t)xcj;
  }
}

// ---------------------------------------------------------------------------
// Final: out = project(project(mu) + o*0.1)
// ---------------------------------------------------------------------------
__global__ __launch_bounds__(256) void final_kernel(const float* __restrict__ o,
                                                    float* __restrict__ out,
                                                    int d) {
  __shared__ float red[256];
  const int row = blockIdx.x;
  const int tid = threadIdx.x;
  const float* orow = o + (size_t)row * d;
  const float base0 = sqrtf(1.f + HEPS);
  float sp[4];
  float ss = 0.f;
#pragma unroll
  for (int i = 0; i < 4; ++i) {
    int j = tid + i * 256;
    float y = 0.1f * orow[j] + ((j == 0) ? base0 : 0.f);
    float s = (j == 0) ? 0.f : fminf(fmaxf(y, -8.f), 8.f);
    sp[i] = s;
    ss += s * s;
  }
  red[tid] = ss;
  __syncthreads();
  for (int off = 128; off > 0; off >>= 1) {
    if (tid < off) red[tid] += red[tid + off];
    __syncthreads();
  }
  const float out0 = sqrtf(1.f + red[0] + HEPS);
#pragma unroll
  for (int i = 0; i < 4; ++i) {
    int j = tid + i * 256;
    out[(size_t)row * d + j] = (j == 0) ? out0 : sp[i];
  }
}

__global__ void f32_to_f16_kernel(const float* __restrict__ src,
                                  half_t* __restrict__ dst, int n) {
  int i = blockIdx.x * blockDim.x + threadIdx.x;
  if (i < n) dst[i] = (half_t)src[i];
}

// ---------------------------------------------------------------------------
// WMMA GEMM: out[M,N] = epilogue(A[M,K] @ W[N,K]^T)
// 128x128 tile, 8 waves (256 thr); wave (rb,cb) owns a 32x64 sub-tile:
// 2 A-frags x 4 B-frags = 8 WMMA per BK=32 step with B reuse.
// Double-buffered LDS filled by global_load_async_to_lds_b128.
// act: 0 none, 1 tanh(.+b), 2 dynamic-metric g from tanh(.+b)*0.1
// ---------------------------------------------------------------------------
#define GLD 48   // LDS row stride (halves) for 32-wide K slab; 96B, 16B-aligned

__global__ __launch_bounds__(256) void gemm_wt_kernel(
    const half_t* __restrict__ A, const half_t* __restrict__ W,
    const float* __restrict__ bias, const half_t* __restrict__ scaleBuf,
    half_t* __restrict__ outH, float* __restrict__ outF,
    int M, int N, int K, int act, float postScale) {
  __shared__ half_t As[2][128 * GLD];
  __shared__ half_t Bs[2][128 * GLD];
  const int tid  = threadIdx.x;
  const int wave = tid >> 5;
  const int lane = tid & 31;
  const int rb = wave >> 1;          // 0..3 -> rows 32*rb
  const int cb = wave & 1;           // 0..1 -> cols 64*cb
  const int m0 = blockIdx.y * 128;
  const int n0 = blockIdx.x * 128;

  const v8f zero = {0.f, 0.f, 0.f, 0.f, 0.f, 0.f, 0.f, 0.f};
  v8f acc[2][4];
#pragma unroll
  for (int u = 0; u < 2; ++u)
#pragma unroll
    for (int t = 0; t < 4; ++t) acc[u][t] = zero;

  auto issue_tile = [&](int buf, int kk) {
#pragma unroll
    for (int i = 0; i < 2; ++i) {            // 128x32 halves = 512 uint4 each
      int v = tid + i * 256;
      int row = v >> 2, c8 = (v & 3) << 3;
      async_ld16(&As[buf][row * GLD + c8],
                 A + (size_t)(m0 + row) * K + kk + c8);
      async_ld16(&Bs[buf][row * GLD + c8],
                 W + (size_t)(n0 + row) * K + kk + c8);
    }
  };

  const int steps = K >> 5;
  issue_tile(0, 0);
  async_wait0();
  __syncthreads();

  for (int s = 0; s < steps; ++s) {
    const int cur = s & 1;
    if (s + 1 < steps) issue_tile(cur ^ 1, (s + 1) << 5);

    const half_t* Ab = &As[cur][(32 * rb) * GLD];
    v16h a0 = frag_ld(Ab, GLD);
    v16h a1 = frag_ld(Ab + 16 * GLD, GLD);
#pragma unroll
    for (int t = 0; t < 4; ++t) {
      v16h b = frag_ld(&Bs[cur][(64 * cb + 16 * t) * GLD], GLD);
      acc[0][t] = wmma_f16(a0, b, acc[0][t]);
      acc[1][t] = wmma_f16(a1, b, acc[1][t]);
    }
    async_wait0();                 // next tile landed (overlapped with WMMA)
    __syncthreads();
  }

  // C layout: lane n = lane&15, row m = v + 8*(lane>=16)
  const int rg = lane >> 4;
  const int nc = lane & 15;
#pragma unroll
  for (int u = 0; u < 2; ++u) {
#pragma unroll
    for (int t = 0; t < 4; ++t) {
#pragma unroll
      for (int v = 0; v < 8; ++v) {
        const int gm = m0 + 32 * rb + 16 * u + v + 8 * rg;
        const int gn = n0 + 64 * cb + 16 * t + nc;
        float val = acc[u][t][v];
        if (bias) val += bias[gn];
        if (act == 1) {
          val = tanhf(val);
        } else if (act == 2) {
          val = tanhf(val) * 0.1f;               // epsilon = 0.1
          val = (gn == 0) ? -fabsf(-1.f + val) : fabsf(1.f + val);
        }
        if (scaleBuf) val *= (float)scaleBuf[(size_t)gm * N + gn];
        val *= postScale;
        if (outH) outH[(size_t)gm * N + gn] = (half_t)val;
        if (outF) outF[(size_t)gm * N + gn] = val;
      }
    }
  }
}

// ---------------------------------------------------------------------------
// Flash attention, dh=64. Q pre-scaled by g/sqrt(dh). Block 128 threads
// (4 waves), each wave owns a 16-row strip of the 64-row Q tile.
// K tiles DMA'd via async-to-LDS; V goes through VGPRs (transposed store).
// grid = (T/64, B*H).
// ---------------------------------------------------------------------------
#define ALD 72   // 144B row stride, 16B-aligned

__global__ __launch_bounds__(128) void attn_kernel(
    const half_t* __restrict__ Q, const half_t* __restrict__ Kb,
    const half_t* __restrict__ Vb, half_t* __restrict__ Ob,
    int T, int d, int H) {
  __shared__ half_t Qs[64 * ALD];
  __shared__ half_t Ks[64 * ALD];
  __shared__ half_t Vst[64 * ALD];   // V transposed: [dh][key]
  __shared__ half_t Ps[64 * ALD];    // softmax probs, [q_local][key]

  const int tid = threadIdx.x, wave = tid >> 5, lane = tid & 31;
  const int bh = blockIdx.y;
  const int b = bh / H, h = bh % H;
  const int q0 = blockIdx.x * 64;
  const size_t rowBase = (size_t)b * T;
  const int dh0 = h * 64;

#pragma unroll
  for (int i = 0; i < 4; ++i) {                  // Q tile 64x64 halves (DMA)
    int v = tid + i * 128;
    int r = v >> 3, c8 = (v & 7) << 3;
    async_ld16(&Qs[r * ALD + c8], Q + (rowBase + q0 + r) * d + dh0 + c8);
  }

  const v8f zero = {0.f, 0.f, 0.f, 0.f, 0.f, 0.f, 0.f, 0.f};
  float rm[8], rl[8];
  v8f of[4];
#pragma unroll
  for (int v = 0; v < 8; ++v) { rm[v] = -1e30f; rl[v] = 0.f; }
#pragma unroll
  for (int t = 0; t < 4; ++t) of[t] = zero;
  async_wait0();
  __syncthreads();

  for (int kt = 0; kt < T; kt += 64) {
#pragma unroll
    for (int i = 0; i < 4; ++i) {                // K tile DMA + V transpose
      int v = tid + i * 128;
      int r = v >> 3, c8 = (v & 7) << 3;
      async_ld16(&Ks[r * ALD + c8], Kb + (rowBase + kt + r) * d + dh0 + c8);
      union { uint4 u; half_t hv[8]; } uu;
      uu.u = *(const uint4*)(Vb + (rowBase + kt + r) * d + dh0 + c8);
#pragma unroll
      for (int j = 0; j < 8; ++j) Vst[(c8 + j) * ALD + r] = uu.hv[j];
    }
    async_wait0();
    __syncthreads();

    // S strip (16x64) = Q_strip @ K^T
    v8f s[4];
#pragma unroll
    for (int t = 0; t < 4; ++t) s[t] = zero;
#pragma unroll
    for (int c = 0; c < 2; ++c) {
      v16h a = frag_ld(&Qs[(wave * 16) * ALD + 32 * c], ALD);
#pragma unroll
      for (int t = 0; t < 4; ++t) {
        v16h bf = frag_ld(&Ks[(t * 16) * ALD + 32 * c], ALD);
        s[t] = wmma_f16(a, bf, s[t]);
      }
    }

    // online softmax: row reductions across the 16-lane N-groups
    float alpha[8];
#pragma unroll
    for (int v = 0; v < 8; ++v) {
      float mx = -1e30f;
#pragma unroll
      for (int t = 0; t < 4; ++t) mx = fmaxf(mx, s[t][v]);
#pragma unroll
      for (int msk = 8; msk >= 1; msk >>= 1) mx = fmaxf(mx, __shfl_xor(mx, msk, 32));
      const float mnew = fmaxf(rm[v], mx);
      alpha[v] = expf(rm[v] - mnew);
      float psum = 0.f;
#pragma unroll
      for (int t = 0; t < 4; ++t) {
        float pp = expf(s[t][v] - mnew);
        s[t][v] = pp;
        psum += pp;
      }
#pragma unroll
      for (int msk = 8; msk >= 1; msk >>= 1) psum += __shfl_xor(psum, msk, 32);
      rl[v] = rl[v] * alpha[v] + psum;
      rm[v] = mnew;
    }

    // stage P (C layout -> row-major LDS) for A-fragment reload
    {
      const int rg = lane >> 4, nc = lane & 15;
#pragma unroll
      for (int t = 0; t < 4; ++t)
#pragma unroll
        for (int v = 0; v < 8; ++v)
          Ps[(wave * 16 + v + 8 * rg) * ALD + t * 16 + nc] = (half_t)s[t][v];
    }
    __syncthreads();

#pragma unroll
    for (int t = 0; t < 4; ++t)
#pragma unroll
      for (int v = 0; v < 8; ++v) of[t][v] *= alpha[v];

    // O strip += P_strip @ V
#pragma unroll
    for (int c = 0; c < 2; ++c) {
      v16h a = frag_ld(&Ps[(wave * 16) * ALD + 32 * c], ALD);
#pragma unroll
      for (int t = 0; t < 4; ++t) {
        v16h bf = frag_ld(&Vst[(t * 16) * ALD + 32 * c], ALD);
        of[t] = wmma_f16(a, bf, of[t]);
      }
    }
    __syncthreads();
  }

  const int rg = lane >> 4, nc = lane & 15;
#pragma unroll
  for (int t = 0; t < 4; ++t)
#pragma unroll
    for (int v = 0; v < 8; ++v) {
      const float val = of[t][v] / rl[v];
      const int gr = q0 + wave * 16 + v + 8 * rg;
      Ob[(rowBase + gr) * d + dh0 + t * 16 + nc] = (half_t)val;
    }
}

// ---------------------------------------------------------------------------
extern "C" void kernel_launch(void* const* d_in, const int* in_sizes, int n_in,
                              void* d_out, int out_size, void* d_ws, size_t ws_size,
                              hipStream_t stream) {
  (void)in_sizes; (void)n_in; (void)out_size; (void)ws_size;
  const float* x  = (const float*)d_in[0];
  const float* W1 = (const float*)d_in[1];
  const float* b1 = (const float*)d_in[2];
  const float* W2 = (const float*)d_in[3];
  const float* b2 = (const float*)d_in[4];
  const float* Wq = (const float*)d_in[5];
  const float* Wk = (const float*)d_in[6];
  const float* Wv = (const float*)d_in[7];
  const float* Wo = (const float*)d_in[8];
  float* out = (float*)d_out;

  const int B = 2, T = 2048, d = 1024, H = 16;
  const int M = B * T;       // 4096
  const int d2 = 2 * d;      // 2048

  char* p = (char*)d_ws;
  auto alloc = [&](size_t bytes) {
    char* r = p;
    p += (bytes + 255) & ~(size_t)255;
    return r;
  };
  half_t* Wq16 = (half_t*)alloc((size_t)d * d * 2);
  half_t* Wk16 = (half_t*)alloc((size_t)d * d * 2);
  half_t* Wv16 = (half_t*)alloc((size_t)d * d * 2);
  half_t* Wo16 = (half_t*)alloc((size_t)d * d * 2);
  half_t* W116 = (half_t*)alloc((size_t)d2 * d * 2);
  half_t* W216 = (half_t*)alloc((size_t)d * d2 * 2);
  half_t* v16b = (half_t*)alloc((size_t)M * d * 2);
  half_t* xc16 = (half_t*)alloc((size_t)M * d * 2);
  half_t* h116 = (half_t*)alloc((size_t)M * d2 * 2);
  half_t* g16  = (half_t*)alloc((size_t)M * d * 2);
  half_t* q16  = (half_t*)alloc((size_t)M * d * 2);
  half_t* k16  = (half_t*)alloc((size_t)M * d * 2);
  half_t* vv16 = (half_t*)alloc((size_t)M * d * 2);
  half_t* at16 = (half_t*)alloc((size_t)M * d * 2);
  float*  o32  = (float*)alloc((size_t)M * d * 4);

  // f16 weight copies
  f32_to_f16_kernel<<<(d * d + 255) / 256, 256, 0, stream>>>(Wq, Wq16, d * d);
  f32_to_f16_kernel<<<(d * d + 255) / 256, 256, 0, stream>>>(Wk, Wk16, d * d);
  f32_to_f16_kernel<<<(d * d + 255) / 256, 256, 0, stream>>>(Wv, Wv16, d * d);
  f32_to_f16_kernel<<<(d * d + 255) / 256, 256, 0, stream>>>(Wo, Wo16, d * d);
  f32_to_f16_kernel<<<(d2 * d + 255) / 256, 256, 0, stream>>>(W1, W116, d2 * d);
  f32_to_f16_kernel<<<(d * d2 + 255) / 256, 256, 0, stream>>>(W2, W216, d * d2);

  // Lorentz log-map + x-mu
  prep_kernel<<<M, 256, 0, stream>>>(x, v16b, xc16, d);

  // metric MLP: h1 = tanh(xc@W1^T+b1); g = metric(tanh(h1@W2^T+b2)*0.1)
  gemm_wt_kernel<<<dim3(d2 / 128, M / 128), 256, 0, stream>>>(
      xc16, W116, b1, nullptr, h116, nullptr, M, d2, d, 1, 1.f);
  gemm_wt_kernel<<<dim3(d / 128, M / 128), 256, 0, stream>>>(
      h116, W216, b2, nullptr, g16, nullptr, M, d, d2, 2, 1.f);

  // q = (v@Wq^T) * g * (1/sqrt(dh)); k = v@Wk^T; vv = v@Wv^T
  gemm_wt_kernel<<<dim3(d / 128, M / 128), 256, 0, stream>>>(
      v16b, Wq16, nullptr, g16, q16, nullptr, M, d, d, 0, 0.125f);
  gemm_wt_kernel<<<dim3(d / 128, M / 128), 256, 0, stream>>>(
      v16b, Wk16, nullptr, nullptr, k16, nullptr, M, d, d, 0, 1.f);
  gemm_wt_kernel<<<dim3(d / 128, M / 128), 256, 0, stream>>>(
      v16b, Wv16, nullptr, nullptr, vv16, nullptr, M, d, d, 0, 1.f);

  // flash attention per (b,h)
  attn_kernel<<<dim3(T / 64, B * H), 128, 0, stream>>>(
      q16, k16, vv16, at16, T, d, H);

  // output projection + final hyperbolic projection
  gemm_wt_kernel<<<dim3(d / 128, M / 128), 256, 0, stream>>>(
      at16, Wo16, nullptr, nullptr, nullptr, o32, M, d, d, 0, 1.f);
  final_kernel<<<M, 256, 0, stream>>>(o32, out, d);
}
